// LocalPointTransformer_14018773254175
// MI455X (gfx1250) — compile-verified, matrix operand order checked by reading, not run on hardware
//
#include <hip/hip_runtime.h>

#define BB 2
#define CC 256
#define NN 4096
#define KNB 16
#define HH 4
#define EPSBN 1e-5f

typedef __attribute__((ext_vector_type(16))) _Float16 v16h;
typedef __attribute__((ext_vector_type(8)))  _Float16 v8h;
typedef __attribute__((ext_vector_type(4)))  _Float16 v4h;
typedef __attribute__((ext_vector_type(8)))  float    v8f;
typedef __attribute__((ext_vector_type(4)))  unsigned int u32x4;
typedef __attribute__((ext_vector_type(8)))  int i32x8;
typedef __attribute__((ext_vector_type(4)))  int i32x4;

#if defined(__has_builtin)
#if __has_builtin(__builtin_amdgcn_tensor_load_to_lds)
#define USE_TDM 1
#else
#define USE_TDM 0
#endif
#else
#define USE_TDM 0
#endif

#if USE_TDM
// ---------------------------------------------------------------------------
// TDM 2D tile load: global (row-major, 2-byte elems) -> LDS, per ISA cdna5 §8.
// D# group0: [1:0]=count=1, [63:32]=lds_addr, [120:64]=global_addr, type=2.
// D# group1: data_size=1(2B), tensor_dim0=rowLen, tensor_dim1=numRows,
//            tile_dim0=tileW, tile_dim1=tileH, tensor_dim0_stride=rowLen.
// ---------------------------------------------------------------------------
__device__ __forceinline__ void tdm_load_2d(unsigned lds_addr, const _Float16* gptr,
                                            int rowLen, int numRows, int tileW, int tileH) {
  const unsigned long long ga = (unsigned long long)(size_t)gptr;
  u32x4 g0;
  g0[0] = 1u;                                              // count=1, user D#
  g0[1] = lds_addr;                                        // LDS byte address
  g0[2] = (unsigned)ga;                                    // global_addr[31:0]
  g0[3] = (unsigned)((ga >> 32) & 0x1FFFFFFull) | (2u << 30); // addr[56:32] | type=2
  i32x8 g1;
  g1[0] = 1 << 16;                                         // data_size=2B, mask=0
  g1[1] = (rowLen & 0xFFFF) << 16;                         // tensor_dim0[15:0]
  g1[2] = (((unsigned)rowLen >> 16) & 0xFFFF) | ((numRows & 0xFFFF) << 16);
  g1[3] = (((unsigned)numRows >> 16) & 0xFFFF) | ((tileW & 0xFFFF) << 16);
  g1[4] = tileH & 0xFFFF;                                  // tile_dim1, tile_dim2=0
  g1[5] = rowLen;                                          // dim0_stride[31:0]
  g1[6] = 0;
  g1[7] = 0;
  const i32x4 z4 = {0, 0, 0, 0};
#if defined(__clang_major__) && (__clang_major__ >= 23)
  const i32x8 z8 = {0, 0, 0, 0, 0, 0, 0, 0};
  __builtin_amdgcn_tensor_load_to_lds(g0, g1, z4, z4, z8, 0);
#else
  __builtin_amdgcn_tensor_load_to_lds(g0, g1, z4, z4, 0);
#endif
}
#endif

// ---------------------------------------------------------------------------
// f32 -> f16 conversion (vectorized). count multiple of 4.
// ---------------------------------------------------------------------------
__global__ void cvt_f16_kernel(const float* __restrict__ src, _Float16* __restrict__ dst,
                               int count) {
  const int i = (blockIdx.x * blockDim.x + threadIdx.x) * 4;
  if (i >= count) return;
  const float4 v = *(const float4*)(src + i);
  v4h o;
  o[0] = (_Float16)v.x; o[1] = (_Float16)v.y; o[2] = (_Float16)v.z; o[3] = (_Float16)v.w;
  *(v4h*)(dst + i) = o;
}

// ---------------------------------------------------------------------------
// Transpose-convert: x [B,C,N] f32 -> xT [B,N,C] f16 via 32x32 LDS tile.
// ---------------------------------------------------------------------------
__global__ void cvt_transpose_kernel(const float* __restrict__ src, _Float16* __restrict__ dst) {
  __shared__ float tile[32][33];
  const int b = blockIdx.z;
  const int c0 = blockIdx.y * 32;
  const int n0 = blockIdx.x * 32;
  const int tx = threadIdx.x;          // 32
  const int ty = threadIdx.y;          // 8
#pragma unroll
  for (int j = 0; j < 4; ++j) {
    const int r = ty + j * 8;
    tile[r][tx] = src[b * (CC * NN) + (c0 + r) * NN + n0 + tx];
  }
  __syncthreads();
#pragma unroll
  for (int j = 0; j < 4; ++j) {
    const int r = ty + j * 8;
    dst[(b * NN + n0 + r) * CC + c0 + tx] = (_Float16)tile[tx][r];
  }
}

// ---------------------------------------------------------------------------
// KNN: brute force; all 4096x3 coords staged in LDS (48KB of 320KB/WGP).
// ---------------------------------------------------------------------------
__global__ void knn_kernel(const float* __restrict__ xyz, int* __restrict__ idx) {
  __shared__ float P[3][NN];
  const int b = blockIdx.y;
  const float* base = xyz + b * 3 * NN;
  for (int m = threadIdx.x; m < NN; m += blockDim.x) {
    P[0][m] = base[m];
    P[1][m] = base[NN + m];
    P[2][m] = base[2 * NN + m];
  }
  __syncthreads();
  const int n = blockIdx.x * blockDim.x + threadIdx.x;
  const float qx = P[0][n], qy = P[1][n], qz = P[2][n];
  float bd[KNB]; int bi[KNB];
#pragma unroll
  for (int t = 0; t < KNB; ++t) { bd[t] = 3.4e38f; bi[t] = 0; }
  for (int m = 0; m < NN; ++m) {
    const float dx = qx - P[0][m], dy = qy - P[1][m], dz = qz - P[2][m];
    const float d = dx * dx + dy * dy + dz * dz;
    if (d < bd[KNB - 1]) {
      bd[KNB - 1] = d; bi[KNB - 1] = m;
#pragma unroll
      for (int t = KNB - 1; t > 0; --t) {
        if (bd[t] < bd[t - 1]) {
          const float td = bd[t]; bd[t] = bd[t - 1]; bd[t - 1] = td;
          const int   ti = bi[t]; bi[t] = bi[t - 1]; bi[t - 1] = ti;
        }
      }
    }
  }
  int* op = idx + (b * NN + n) * KNB;
#pragma unroll
  for (int t = 0; t < KNB; ++t) op[t] = bi[t];
}

// ---------------------------------------------------------------------------
// w2sum[h][c] = sum_d rp_w2[h*64+d][c]
// ---------------------------------------------------------------------------
__global__ void w2sum_kernel(const float* __restrict__ rp_w2, float* __restrict__ w2s) {
  const int c = blockIdx.x * blockDim.x + threadIdx.x;
  if (c >= CC) return;
#pragma unroll
  for (int h = 0; h < HH; ++h) {
    float s = 0.f;
    for (int d = 0; d < CC / HH; ++d) s += rp_w2[(h * (CC / HH) + d) * CC + c];
    w2s[h * CC + c] = s;
  }
}

// ---------------------------------------------------------------------------
// Relative-position bias: biasb[b,n,k,h] = w2sum[h,:] . relu(bn(rp_w1 @ rel))
// ---------------------------------------------------------------------------
__global__ void rpbias_kernel(const float* __restrict__ xyz, const int* __restrict__ idx,
                              const float* __restrict__ rp_w1,
                              const float* __restrict__ rp_g, const float* __restrict__ rp_b,
                              const float* __restrict__ rp_m, const float* __restrict__ rp_v,
                              const float* __restrict__ w2s, float* __restrict__ biasb) {
  const int t = blockIdx.x * blockDim.x + threadIdx.x;
  if (t >= BB * NN * KNB) return;
  const int k = t % KNB;
  const int n = (t / KNB) % NN;
  const int b = t / (KNB * NN);
  const int nb = idx[(b * NN + n) * KNB + k];
  const float* pb = xyz + b * 3 * NN;
  const float rx = pb[nb] - pb[n];
  const float ry = pb[NN + nb] - pb[NN + n];
  const float rz = pb[2 * NN + nb] - pb[2 * NN + n];
  float acc0 = 0.f, acc1 = 0.f, acc2 = 0.f, acc3 = 0.f;
  for (int c = 0; c < CC; ++c) {
    float h1 = rp_w1[c * 3] * rx + rp_w1[c * 3 + 1] * ry + rp_w1[c * 3 + 2] * rz;
    const float s = rp_g[c] * rsqrtf(rp_v[c] + EPSBN);
    h1 = (h1 - rp_m[c]) * s + rp_b[c];
    h1 = fmaxf(h1, 0.f);
    acc0 += w2s[c] * h1;
    acc1 += w2s[CC + c] * h1;
    acc2 += w2s[2 * CC + c] * h1;
    acc3 += w2s[3 * CC + c] * h1;
  }
  float* op = biasb + ((b * NN + n) * KNB + k) * HH;
  op[0] = acc0; op[1] = acc1; op[2] = acc2; op[3] = acc3;
}

// ---------------------------------------------------------------------------
// Fused attention: logits -> softmax(K=16) -> attn @ v. One thread per
// (b,n,h). Output written [N,C]-contiguous in f16 (K-major for next GEMM).
// ---------------------------------------------------------------------------
__global__ void attn_kernel(const float* __restrict__ qbuf, const float* __restrict__ kvT,
                            const int* __restrict__ idx, const float* __restrict__ biasb,
                            _Float16* __restrict__ attT) {
  const int t = blockIdx.x * blockDim.x + threadIdx.x;
  if (t >= BB * NN * HH) return;
  const int h = t % HH;
  const int n = (t / HH) % NN;
  const int b = t / (HH * NN);
  const int D = CC / HH;                 // 64
  const float scale = 0.125f;            // d^-0.5

  const int qb = b * CC * NN;
  float qreg[64];
#pragma unroll
  for (int d = 0; d < 64; ++d) qreg[d] = qbuf[qb + (h * D + d) * NN + n];

  int koff[KNB], voff[KNB];
#pragma unroll
  for (int k = 0; k < KNB; ++k) {
    const int nb = idx[(b * NN + n) * KNB + k];
    koff[k] = ((b * NN + nb) * (2 * CC) + h * D) >> 2;      // float4 index
    voff[k] = koff[k] + (CC >> 2);
  }

  const float4* kv4 = (const float4*)kvT;
  float lg[KNB];
#pragma unroll
  for (int k = 0; k < KNB; ++k) {
    float dot = 0.f;
#pragma unroll
    for (int d4 = 0; d4 < 16; ++d4) {
      const float4 kk = kv4[koff[k] + d4];
      dot += qreg[4 * d4] * kk.x + qreg[4 * d4 + 1] * kk.y +
             qreg[4 * d4 + 2] * kk.z + qreg[4 * d4 + 3] * kk.w;
    }
    lg[k] = dot * scale + biasb[((b * NN + n) * KNB + k) * HH + h];
  }
  float mx = lg[0];
#pragma unroll
  for (int k = 1; k < KNB; ++k) mx = fmaxf(mx, lg[k]);
  float sum = 0.f;
#pragma unroll
  for (int k = 0; k < KNB; ++k) { lg[k] = __expf(lg[k] - mx); sum += lg[k]; }
  const float inv = 1.f / sum;
#pragma unroll
  for (int k = 0; k < KNB; ++k) lg[k] *= inv;

  _Float16* op = attT + (b * NN + n) * CC + h * D;
  for (int d4 = 0; d4 < 16; ++d4) {
    float a0 = 0.f, a1 = 0.f, a2 = 0.f, a3 = 0.f;
#pragma unroll
    for (int k = 0; k < KNB; ++k) {
      const float4 vv = kv4[voff[k] + d4];
      a0 += lg[k] * vv.x; a1 += lg[k] * vv.y; a2 += lg[k] * vv.z; a3 += lg[k] * vv.w;
    }
    v4h o;
    o[0] = (_Float16)a0; o[1] = (_Float16)a1; o[2] = (_Float16)a2; o[3] = (_Float16)a3;
    *(v4h*)(op + d4 * 4) = o;
  }
}

// ---------------------------------------------------------------------------
// WMMA GEMM: Out = epilogue(Wh[M x Kd] @ XT^T) per batch; XT is K-major
// [N x Kd] f16, so BOTH LDS tiles are plain 2D row-tile copies.
// 256 threads = 8 wave32 waves; block tile 128x64; wave w owns rows
// [16w,16w+16), 4 wmma tiles per K-step (A-fragment reuse), 4x v8f acc.
// Staging: TDM tensor_load_to_lds (wave 0 issues, TENSORcnt-synced) with a
// ds_store_b128 copy fallback; double-buffered LDS, one barrier per K-step.
// f16 secondary output (OH) is written K-major [N x M] for the next GEMM.
// ---------------------------------------------------------------------------
__global__ void gemm_wmma_f16(const _Float16* __restrict__ Wh, const _Float16* __restrict__ XTh,
                              float* __restrict__ OutF, _Float16* __restrict__ OutH,
                              int M, int Kd, int N,
                              int xStride, int oStride, int ldOut, int transpose,
                              const float* __restrict__ bias,
                              const float* __restrict__ res, int rStride,
                              const float* __restrict__ bng, const float* __restrict__ bnb,
                              const float* __restrict__ bnm, const float* __restrict__ bnv,
                              int relu) {
  __shared__ __align__(16) _Float16 Wa[2][128 * 32];
  __shared__ __align__(16) _Float16 Xt[2][64 * 32];   // [n][k], K-major

  const int b = blockIdx.z;
  const _Float16* Xb = XTh + b * xStride;
  const int m0 = blockIdx.y * 128;
  const int n0 = blockIdx.x * 64;
  const int tid = threadIdx.x;
  const int wave = tid >> 5;
  const int lane = tid & 31;
  const int row = lane & 15;
  const int hs = lane >> 4;

  v8f acc[4] = {{}, {}, {}, {}};

#if USE_TDM
  auto stage = [&](int p, int k0) {
    if (wave == 0) {
      tdm_load_2d((unsigned)(size_t)&Wa[p][0], Wh + m0 * Kd + k0, Kd, M, 32, 128);
      tdm_load_2d((unsigned)(size_t)&Xt[p][0], Xb + n0 * Kd + k0, Kd, N, 32, 64);
    }
  };
  auto stage_wait = [&]() {
    if (wave == 0) __builtin_amdgcn_s_wait_tensorcnt(0);
  };
#else
  const int swr = tid >> 1, swc = (tid & 1) * 16;   // W: 16 halves/thread
  const int sxn = tid >> 2, sxc = (tid & 3) * 8;    // X: 8 halves/thread
  auto stage = [&](int p, int k0) {
    const _Float16* ws = Wh + (m0 + swr) * Kd + k0 + swc;
    *(v8h*)&Wa[p][swr * 32 + swc]     = *(const v8h*)ws;
    *(v8h*)&Wa[p][swr * 32 + swc + 8] = *(const v8h*)(ws + 8);
    *(v8h*)&Xt[p][sxn * 32 + sxc] = *(const v8h*)(Xb + (n0 + sxn) * Kd + k0 + sxc);
  };
  auto stage_wait = [&]() {};
#endif

  stage(0, 0);
  stage_wait();
  __syncthreads();
  int p = 0;
  for (int k0 = 0; k0 < Kd; k0 += 32) {
    const bool more = (k0 + 32) < Kd;
    if (more) stage(p ^ 1, k0 + 32);

    const _Float16* ap = &Wa[p][(wave * 16 + row) * 32 + hs * 8];
    const v8h alo = *(const v8h*)ap;
    const v8h ahi = *(const v8h*)(ap + 16);
    const v16h a = __builtin_shufflevector(alo, ahi, 0,1,2,3,4,5,6,7,8,9,10,11,12,13,14,15);
#pragma unroll
    for (int tc = 0; tc < 4; ++tc) {
      const _Float16* bp = &Xt[p][(tc * 16 + row) * 32 + hs * 8];
      const v8h blo = *(const v8h*)bp;
      const v8h bhi = *(const v8h*)(bp + 16);
      const v16h bf = __builtin_shufflevector(blo, bhi, 0,1,2,3,4,5,6,7,8,9,10,11,12,13,14,15);
      acc[tc] = __builtin_amdgcn_wmma_f32_16x16x32_f16(false, a, false, bf, (short)0,
                                                       acc[tc], false, false);
    }
    if (more) stage_wait();
    __syncthreads();
    p ^= 1;
  }

  // C/D layout: VGPR r on lane L -> M = r + 8*(L>=16), N = L&15 (within tile)
  float* OF = OutF ? OutF + b * oStride : (float*)nullptr;
  _Float16* OH = OutH ? OutH + b * oStride : (_Float16*)nullptr;
  const float* Rb = res ? res + b * rStride : (const float*)nullptr;
#pragma unroll
  for (int tc = 0; tc < 4; ++tc) {
    const int n = n0 + tc * 16 + row;
#pragma unroll
    for (int r = 0; r < 8; ++r) {
      const int m = m0 + wave * 16 + hs * 8 + r;
      float v = acc[tc][r];
      if (bias) v += bias[m];
      if (Rb)   v += Rb[m * N + n];
      if (bng) {
        const float s = bng[m] * rsqrtf(bnv[m] + EPSBN);
        v = (v - bnm[m]) * s + bnb[m];
      }
      if (relu) v = fmaxf(v, 0.f);
      if (OF) {
        if (transpose) OF[n * ldOut + m] = v;
        else           OF[m * ldOut + n] = v;
      }
      if (OH) OH[n * M + m] = (_Float16)v;     // K-major for next GEMM
    }
  }
}

// ---------------------------------------------------------------------------
extern "C" void kernel_launch(void* const* d_in, const int* in_sizes, int n_in,
                              void* d_out, int out_size, void* d_ws, size_t ws_size,
                              hipStream_t stream) {
  const float* x     = (const float*)d_in[0];
  const float* xyz   = (const float*)d_in[1];
  const float* Wq    = (const float*)d_in[2];
  const float* Wkv   = (const float*)d_in[3];
  const float* Wproj = (const float*)d_in[4];
  const float* rp_w1 = (const float*)d_in[5];
  const float* rp_g  = (const float*)d_in[6];
  const float* rp_b  = (const float*)d_in[7];
  const float* rp_m  = (const float*)d_in[8];
  const float* rp_v  = (const float*)d_in[9];
  const float* rp_w2 = (const float*)d_in[10];
  const float* bn1_g = (const float*)d_in[11];
  const float* bn1_b = (const float*)d_in[12];
  const float* bn1_m = (const float*)d_in[13];
  const float* bn1_v = (const float*)d_in[14];
  const float* bn2_g = (const float*)d_in[15];
  const float* bn2_b = (const float*)d_in[16];
  const float* bn2_m = (const float*)d_in[17];
  const float* bn2_v = (const float*)d_in[18];
  const float* ffnw1 = (const float*)d_in[19];
  const float* ffnb1 = (const float*)d_in[20];
  const float* ffnw2 = (const float*)d_in[21];
  const float* ffnb2 = (const float*)d_in[22];
  float* out = (float*)d_out;

  const int CN = CC * NN;                       // 1,048,576
  char* ws = (char*)d_ws;
  size_t off = 0;
  int*      idxb  = (int*)(ws + off);      off += (size_t)BB * NN * KNB * 4;
  float*    qbuf  = (float*)(ws + off);    off += (size_t)BB * CN * 4;
  float*    kvT   = (float*)(ws + off);    off += (size_t)BB * NN * 2 * CC * 4;
  float*    w2s   = (float*)(ws + off);    off += (size_t)HH * CC * 4;
  float*    biasb = (float*)(ws + off);    off += (size_t)BB * NN * KNB * HH * 4;
  float*    x1    = (float*)(ws + off);    off += (size_t)BB * CN * 4;
  _Float16* xTh   = (_Float16*)(ws + off); off += (size_t)BB * CN * 2;   // [B,N,C]
  _Float16* attT  = (_Float16*)(ws + off); off += (size_t)BB * CN * 2;   // [B,N,C]
  _Float16* x1T   = (_Float16*)(ws + off); off += (size_t)BB * CN * 2;   // [B,N,C]
  _Float16* hT    = (_Float16*)(ws + off); off += (size_t)BB * 4 * CN * 2; // [B,N,4C]
  _Float16* wqh   = (_Float16*)(ws + off); off += (size_t)CC * CC * 2;
  _Float16* wkvh  = (_Float16*)(ws + off); off += (size_t)2 * CC * CC * 2;
  _Float16* wph   = (_Float16*)(ws + off); off += (size_t)CC * CC * 2;
  _Float16* w1h   = (_Float16*)(ws + off); off += (size_t)4 * CC * CC * 2;
  _Float16* w2h   = (_Float16*)(ws + off); off += (size_t)4 * CC * CC * 2;
  (void)ws_size; (void)in_sizes; (void)n_in; (void)out_size;

  // 0) one-shot conversions: weights f16; x -> K-major f16
  cvt_f16_kernel<<<(CC * CC) / 1024, 256, 0, stream>>>(Wq, wqh, CC * CC);
  cvt_f16_kernel<<<(2 * CC * CC) / 1024, 256, 0, stream>>>(Wkv, wkvh, 2 * CC * CC);
  cvt_f16_kernel<<<(CC * CC) / 1024, 256, 0, stream>>>(Wproj, wph, CC * CC);
  cvt_f16_kernel<<<(4 * CC * CC) / 1024, 256, 0, stream>>>(ffnw1, w1h, 4 * CC * CC);
  cvt_f16_kernel<<<(4 * CC * CC) / 1024, 256, 0, stream>>>(ffnw2, w2h, 4 * CC * CC);
  cvt_transpose_kernel<<<dim3(NN / 32, CC / 32, BB), dim3(32, 8), 0, stream>>>(x, xTh);

  // 1) KNN indices
  knn_kernel<<<dim3(NN / 256, BB), 256, 0, stream>>>(xyz, idxb);

  // 2) collapsed rp_w2 (sum over d per head)
  w2sum_kernel<<<1, 256, 0, stream>>>(rp_w2, w2s);

  // 3) q = Wq @ x -> qbuf [C,N] f32
  gemm_wmma_f16<<<dim3(NN / 64, CC / 128, BB), 256, 0, stream>>>(
      wqh, xTh, qbuf, nullptr, CC, CC, NN, CN, CN, NN, 0,
      nullptr, nullptr, 0, nullptr, nullptr, nullptr, nullptr, 0);

  // 4) kv = Wkv @ x -> kvT [N, 2C] f32 (transposed store)
  gemm_wmma_f16<<<dim3(NN / 64, (2 * CC) / 128, BB), 256, 0, stream>>>(
      wkvh, xTh, kvT, nullptr, 2 * CC, CC, NN, CN, NN * 2 * CC, 2 * CC, 1,
      nullptr, nullptr, 0, nullptr, nullptr, nullptr, nullptr, 0);

  // 5) relative-position bias per (b,n,k,h)
  rpbias_kernel<<<(BB * NN * KNB + 255) / 256, 256, 0, stream>>>(
      xyz, idxb, rp_w1, rp_g, rp_b, rp_m, rp_v, w2s, biasb);

  // 6) fused attention -> attT f16 [N,C] per batch
  attn_kernel<<<(BB * NN * HH + 255) / 256, 256, 0, stream>>>(
      qbuf, kvT, idxb, biasb, attT);

  // 7) x1 = bn1(x + Wproj @ attn) -> f32 x1 (residual) + f16 x1T (K-major)
  gemm_wmma_f16<<<dim3(NN / 64, CC / 128, BB), 256, 0, stream>>>(
      wph, attT, x1, x1T, CC, CC, NN, CN, CN, NN, 0,
      nullptr, x, CN, bn1_g, bn1_b, bn1_m, bn1_v, 0);

  // 8) h = relu(ffn_w1 @ x1 + b1) -> hT f16 [N,4C] (K-major only)
  gemm_wmma_f16<<<dim3(NN / 64, (4 * CC) / 128, BB), 256, 0, stream>>>(
      w1h, x1T, nullptr, hT, 4 * CC, CC, NN, CN, 4 * CN, NN, 0,
      ffnb1, nullptr, 0, nullptr, nullptr, nullptr, nullptr, 1);

  // 9) out = bn2(x1 + ffn_w2 @ h + b2) -> d_out f32
  gemm_wmma_f16<<<dim3(NN / 64, CC / 128, BB), 256, 0, stream>>>(
      w2h, hT, out, nullptr, CC, 4 * CC, NN, 4 * CN, CN, NN, 0,
      ffnb2, x1, CN, bn2_g, bn2_b, bn2_m, bn2_v, 0);
}
